// CausalSelfAttention_75316546503052
// MI455X (gfx1250) — matrix operand order repaired and checked by reading, not compiled, not run on hardware
//
#include <hip/hip_runtime.h>

typedef unsigned short ushort_t;
typedef __attribute__((ext_vector_type(16))) __bf16 bf16x16;
typedef __attribute__((ext_vector_type(8)))  float   f32x8;

#if __has_builtin(__builtin_amdgcn_tensor_load_to_lds) && \
    __has_builtin(__builtin_amdgcn_s_wait_tensorcnt)
#define USE_TDM 1
typedef __attribute__((ext_vector_type(4))) unsigned int u32x4;
typedef __attribute__((ext_vector_type(8))) int          i32x8;
typedef __attribute__((ext_vector_type(4))) int          i32x4;
#endif

// ---------- helpers ----------
__device__ __forceinline__ ushort_t f2bf(float f) {
  union { float f; unsigned u; } v; v.f = f;
  unsigned r = v.u + 0x7FFFu + ((v.u >> 16) & 1u);   // round-to-nearest-even
  return (ushort_t)(r >> 16);
}

union Frag16 { bf16x16 v; uint4 q[2]; };

// A-matrix 16x32 bf16 fragment (ISA 7.12.2): lane L holds row M=L%16;
// elements 0..7 -> K = 8g..8g+7 ; elements 8..15 -> K = 16+8g..23+8g  (g=L/16)
__device__ __forceinline__ bf16x16 frag_a(const ushort_t* row, int g, int k0) {
  Frag16 f;
  f.q[0] = *(const uint4*)(row + k0 + 8 * g);
  f.q[1] = *(const uint4*)(row + k0 + 16 + 8 * g);
  return f.v;
}
// B-matrix 32x16 bf16 fragment: lane L holds col N=L%16;
// elements 0..7 -> K = 8g..8g+7 ; elements 8..15 -> K = 16+8g..23+8g (source stored [N][K]).
__device__ __forceinline__ bf16x16 frag_b(const ushort_t* row, int g, int k0) {
  Frag16 f;
  const ushort_t* p = row + k0 + 8 * g;
  f.q[0] = *(const uint4*)(p);
  f.q[1] = *(const uint4*)(p + 16);
  return f.v;
}

__device__ __forceinline__ f32x8 wmma_bf16(bf16x16 a, bf16x16 b, f32x8 c) {
  return __builtin_amdgcn_wmma_f32_16x16x32_bf16(false, a, false, b, (short)0, c,
                                                 false, false);
}

#ifdef USE_TDM
__device__ __forceinline__ unsigned lds_off(const void* p) {
  return (unsigned)(unsigned long long)p;   // flat addr low 32 bits == LDS byte addr
}
// 2-D tile (64 rows x 64 bf16) -> LDS with HW padding: 16B pad per 128B row
// (reproduces the 72-ushort padded row stride used by the fragment loaders)
__device__ __forceinline__ void tdm_load_tile_2d(unsigned lds_addr,
                                                 const ushort_t* gptr,
                                                 unsigned row_len, unsigned nrows,
                                                 unsigned row_stride) {
  const unsigned long long ga = (unsigned long long)gptr;
  u32x4 g0;
  g0[0] = 1u;                                    // count=1 (valid), user mode
  g0[1] = lds_addr;                              // D#.lds_addr
  g0[2] = (unsigned)ga;                          // global_addr[31:0]
  g0[3] = ((unsigned)(ga >> 32) & 0x01FFFFFFu)   // global_addr[56:32]
          | (2u << 30);                          // type = 2 (image)
  i32x8 g1;
  g1[0] = (int)((1u << 16)      // data_size = 2 bytes
                | (1u << 20)    // pad_enable
                | (4u << 22)    // pad_interval: 32 dwords (= 128B row)
                | (3u << 25));  // pad_amount: 4 dwords (16B)
  g1[1] = (int)(row_len << 16);                       // tensor_dim0 lo
  g1[2] = (int)((row_len >> 16) | (nrows << 16));     // dim0 hi | tensor_dim1 lo
  g1[3] = (int)((nrows >> 16) | (64u << 16));         // dim1 hi | tile_dim0 = 64
  g1[4] = 64;                                         // tile_dim1 = 64, tile_dim2 = 0
  g1[5] = (int)row_stride;                            // tensor_dim0_stride lo32
  g1[6] = 0;
  g1[7] = 0;
  const i32x4 z = {0, 0, 0, 0};
#if __clang_major__ >= 23
  const i32x8 z8 = {0, 0, 0, 0, 0, 0, 0, 0};
  __builtin_amdgcn_tensor_load_to_lds(g0, g1, z, z, z8, 0);
#else
  __builtin_amdgcn_tensor_load_to_lds(g0, g1, z, z, 0);
#endif
}
#endif

// ---------- conversion kernels ----------
__global__ __launch_bounds__(256) void k_cvt_bf16(const float* __restrict__ s,
                                                  ushort_t* __restrict__ d, int n) {
  int i = blockIdx.x * 256 + threadIdx.x;
  if (i < n) d[i] = f2bf(s[i]);
}

// d[n*K + k] = s[k*N + n]   (store weights [N][K] so K is contiguous)
__global__ __launch_bounds__(256) void k_cvt_tr(const float* __restrict__ s,
                                                ushort_t* __restrict__ d, int K, int N) {
  int i = blockIdx.x * 256 + threadIdx.x;
  if (i < K * N) {
    int k = i % K, n = i / K;
    d[i] = f2bf(s[k * N + n]);
  }
}

// ---------- QKV GEMM: 256x64 block tile, 64x32 wave tile (8 WMMA / k-step) ----------
__global__ __launch_bounds__(256) void k_gemm_qkv(
    const ushort_t* __restrict__ xb, const ushort_t* __restrict__ wab,
    const float* __restrict__ b_attn, ushort_t* __restrict__ qb,
    ushort_t* __restrict__ kb, ushort_t* __restrict__ vtb) {
  constexpr int AST = 40, BST = 40;
  __shared__ ushort_t lA[256 * AST];
  __shared__ ushort_t lB[64 * BST];
  const int tid = threadIdx.x, lane = tid & 31, w = tid >> 5;
  const int wm = w >> 1, wn = w & 1;
  const int g = lane >> 4, nn = lane & 15;
  const int m0 = blockIdx.x * 256;
  const int by = blockIdx.y, n0 = by * 64;

  f32x8 acc[4][2] = {};

  for (int k0 = 0; k0 < 1024; k0 += 32) {
    __syncthreads();
    {  // A: one 64B row per thread
      const uint4* s = (const uint4*)(xb + (m0 + tid) * 1024 + k0);
      uint4* d = (uint4*)(lA + tid * AST);
      d[0] = s[0]; d[1] = s[1]; d[2] = s[2]; d[3] = s[3];
    }
    if (tid < 64) {  // B: one 64B row per thread
      const uint4* s = (const uint4*)(wab + (n0 + tid) * 1024 + k0);
      uint4* d = (uint4*)(lB + tid * BST);
      d[0] = s[0]; d[1] = s[1]; d[2] = s[2]; d[3] = s[3];
    }
    __syncthreads();
    bf16x16 b0 = frag_b(lB + (wn * 32 + nn) * BST, g, 0);
    bf16x16 b1 = frag_b(lB + (wn * 32 + 16 + nn) * BST, g, 0);
#pragma unroll
    for (int im = 0; im < 4; ++im) {
      bf16x16 a = frag_a(lA + (wm * 64 + im * 16 + nn) * AST, g, 0);
      acc[im][0] = wmma_bf16(a, b0, acc[im][0]);
      acc[im][1] = wmma_bf16(a, b1, acc[im][1]);
    }
  }

  const int sec = by >> 4;  // 0:q 1:k 2:v  (N-tile = 64 = head size)
  const int h = by & 15;
#pragma unroll
  for (int im = 0; im < 4; ++im)
#pragma unroll
    for (int in = 0; in < 2; ++in) {
      const int d = wn * 32 + in * 16 + nn;  // 0..63 within head
      const float bias = b_attn[n0 + d];
#pragma unroll
      for (int r = 0; r < 8; ++r) {
        const int m = m0 + wm * 64 + im * 16 + r + 8 * g;
        const int b = m >> 11, t = m & 2047;
        const ushort_t hv = f2bf(acc[im][in][r] + bias);
        if (sec == 0)      qb[(((b * 16 + h) * 2048 + t) << 6) + d] = hv;
        else if (sec == 1) kb[(((b * 16 + h) * 2048 + t) << 6) + d] = hv;
        else               vtb[(((b * 16 + h) * 64 + d) << 11) + t] = hv;
      }
    }
}

// ---------- flash attention: per (b,h), 128 queries per block ----------
__global__ __launch_bounds__(256) void k_attn(const ushort_t* __restrict__ qb,
                                              const ushort_t* __restrict__ kbuf,
                                              const ushort_t* __restrict__ vtb,
                                              ushort_t* __restrict__ yb) {
  constexpr int KST = 72, VST = 72, PST = 72;
  __shared__ ushort_t lK[2][64 * KST];     // [key][hs], double-buffered
  __shared__ ushort_t lV[2][64 * VST];     // [d][key], double-buffered
  __shared__ ushort_t lP[8 * 16 * PST];    // per-wave P strip [16 q][64 key]
  const int tid = threadIdx.x, lane = tid & 31, w = tid >> 5;
  const int g = lane >> 4, nn = lane & 15;
  const int qt = blockIdx.x, h = blockIdx.y, b = blockIdx.z;
  const ushort_t* qh = qb + (size_t)((b * 16 + h) * 2048) * 64;
  const ushort_t* kh = kbuf + (size_t)((b * 16 + h) * 2048) * 64;
  const ushort_t* vh = vtb + (size_t)((b * 16 + h) * 64) * 2048;
  const int qbase = qt * 128 + w * 16;

  bf16x16 aq[2];
  {
    const ushort_t* qrow = qh + (qbase + nn) * 64;
    aq[0] = frag_a(qrow, g, 0);
    aq[1] = frag_a(qrow, g, 32);
  }

  f32x8 o[4] = {};
  float mrow[8], lrow[8];
#pragma unroll
  for (int r = 0; r < 8; ++r) { mrow[r] = -3.0e38f; lrow[r] = 0.f; }

  const int nkb = 2 * qt + 2;  // causal: keys up to qt*128+127

#ifdef USE_TDM
  if (w == 0) {   // prefetch key-block 0 via Tensor Data Mover
    tdm_load_tile_2d(lds_off(&lK[0][0]), kh, 64, 64, 64);
    tdm_load_tile_2d(lds_off(&lV[0][0]), vh, 64, 64, 2048);
  }
#else
  const int ldrow = tid >> 2, ldc = (tid & 3) * 16;
#endif

  for (int kb = 0; kb < nkb; ++kb) {
    const int kbase = kb * 64;
#ifdef USE_TDM
    const int cur = kb & 1;
    if (w == 0) __builtin_amdgcn_s_wait_tensorcnt(0);  // buffer `cur` landed
    __syncthreads();
    if (w == 0 && (kb + 1) < nkb) {   // overlap next block's DMA with compute
      const int nb = cur ^ 1;
      const int nk = kbase + 64;
      tdm_load_tile_2d(lds_off(&lK[nb][0]), kh + nk * 64, 64, 64, 64);
      tdm_load_tile_2d(lds_off(&lV[nb][0]), vh + nk, 64, 64, 2048);
    }
#else
    const int cur = 0;
    __syncthreads();
    {
      const uint4* s = (const uint4*)(kh + (kbase + ldrow) * 64 + ldc);
      uint4* d = (uint4*)(&lK[0][0] + ldrow * KST + ldc);
      d[0] = s[0]; d[1] = s[1];
    }
    {
      const uint4* s = (const uint4*)(vh + ldrow * 2048 + kbase + ldc);
      uint4* d = (uint4*)(&lV[0][0] + ldrow * VST + ldc);
      d[0] = s[0]; d[1] = s[1];
    }
    __syncthreads();
#endif
    const ushort_t* lKc = &lK[cur][0];
    const ushort_t* lVc = &lV[cur][0];

    // S = Q K^T  (4 key sub-frags x 2 k-steps)
    f32x8 s4[4];
#pragma unroll
    for (int j = 0; j < 4; ++j) {
      f32x8 sj = {};
      const ushort_t* krow = lKc + (j * 16 + nn) * KST;
      sj = wmma_bf16(aq[0], frag_b(krow, g, 0), sj);
      sj = wmma_bf16(aq[1], frag_b(krow, g, 32), sj);
      s4[j] = sj;
    }

    // scale + causal mask + row max (row m = r + 8g, col = lane%16)
    float mx[8];
#pragma unroll
    for (int r = 0; r < 8; ++r) {
      const int qi = qbase + r + 8 * g;
      float m = -3.0e38f;
#pragma unroll
      for (int j = 0; j < 4; ++j) {
        const int ki = kbase + j * 16 + nn;
        float sv = s4[j][r] * 0.125f;
        sv = (ki <= qi) ? sv : -3.0e38f;
        s4[j][r] = sv;
        m = fmaxf(m, sv);
      }
      mx[r] = m;
    }
#pragma unroll
    for (int off = 8; off; off >>= 1)
#pragma unroll
      for (int r = 0; r < 8; ++r)
        mx[r] = fmaxf(mx[r], __shfl_xor(mx[r], off, 32));

    float al[8];
#pragma unroll
    for (int r = 0; r < 8; ++r) {
      const float nm = fmaxf(mrow[r], mx[r]);
      al[r] = __expf(mrow[r] - nm);
      mrow[r] = nm;
    }
    float rs[8];
#pragma unroll
    for (int r = 0; r < 8; ++r) {
      float s = 0.f;
#pragma unroll
      for (int j = 0; j < 4; ++j) {
        const float p = __expf(s4[j][r] - mrow[r]);
        s4[j][r] = p;
        s += p;
      }
      rs[r] = s;
    }
#pragma unroll
    for (int off = 8; off; off >>= 1)
#pragma unroll
      for (int r = 0; r < 8; ++r) rs[r] += __shfl_xor(rs[r], off, 32);
#pragma unroll
    for (int r = 0; r < 8; ++r) lrow[r] = lrow[r] * al[r] + rs[r];
#pragma unroll
    for (int j = 0; j < 4; ++j)
#pragma unroll
      for (int r = 0; r < 8; ++r) o[j][r] *= al[r];

    // P -> per-wave LDS strip, re-enter as A fragment
    ushort_t* pw = lP + w * 16 * PST;
#pragma unroll
    for (int j = 0; j < 4; ++j)
#pragma unroll
      for (int r = 0; r < 8; ++r)
        pw[(r + 8 * g) * PST + j * 16 + nn] = f2bf(s4[j][r]);

#pragma unroll
    for (int ks = 0; ks < 2; ++ks) {
      bf16x16 ap = frag_a(pw + nn * PST, g, ks * 32);
#pragma unroll
      for (int j = 0; j < 4; ++j) {
        const ushort_t* vrow = lVc + (j * 16 + nn) * VST;
        o[j] = wmma_bf16(ap, frag_b(vrow, g, ks * 32), o[j]);
      }
    }
  }

  // epilogue: y[b,t,h*64+d] = o / l
#pragma unroll
  for (int r = 0; r < 8; ++r) {
    const int t = qbase + r + 8 * g;
    const float inv = 1.0f / lrow[r];
#pragma unroll
    for (int j = 0; j < 4; ++j)
      yb[(size_t)(b * 2048 + t) * 1024 + h * 64 + j * 16 + nn] = f2bf(o[j][r] * inv);
  }
}

// ---------- projection GEMM: out = y @ w_proj + b_proj (f32 out) ----------
__global__ __launch_bounds__(256) void k_gemm_proj(const ushort_t* __restrict__ yb,
                                                   const ushort_t* __restrict__ wpb,
                                                   const float* __restrict__ b_proj,
                                                   float* __restrict__ out) {
  constexpr int AST = 40, BST = 40;
  __shared__ ushort_t lA[256 * AST];
  __shared__ ushort_t lB[64 * BST];
  const int tid = threadIdx.x, lane = tid & 31, w = tid >> 5;
  const int wm = w >> 1, wn = w & 1;
  const int g = lane >> 4, nn = lane & 15;
  const int m0 = blockIdx.x * 256, n0 = blockIdx.y * 64;

  f32x8 acc[4][2] = {};

  for (int k0 = 0; k0 < 1024; k0 += 32) {
    __syncthreads();
    {
      const uint4* s = (const uint4*)(yb + (m0 + tid) * 1024 + k0);
      uint4* d = (uint4*)(lA + tid * AST);
      d[0] = s[0]; d[1] = s[1]; d[2] = s[2]; d[3] = s[3];
    }
    if (tid < 64) {
      const uint4* s = (const uint4*)(wpb + (n0 + tid) * 1024 + k0);
      uint4* d = (uint4*)(lB + tid * BST);
      d[0] = s[0]; d[1] = s[1]; d[2] = s[2]; d[3] = s[3];
    }
    __syncthreads();
    bf16x16 b0 = frag_b(lB + (wn * 32 + nn) * BST, g, 0);
    bf16x16 b1 = frag_b(lB + (wn * 32 + 16 + nn) * BST, g, 0);
#pragma unroll
    for (int im = 0; im < 4; ++im) {
      bf16x16 a = frag_a(lA + (wm * 64 + im * 16 + nn) * AST, g, 0);
      acc[im][0] = wmma_bf16(a, b0, acc[im][0]);
      acc[im][1] = wmma_bf16(a, b1, acc[im][1]);
    }
  }

#pragma unroll
  for (int im = 0; im < 4; ++im)
#pragma unroll
    for (int in = 0; in < 2; ++in) {
      const int col = n0 + wn * 32 + in * 16 + nn;
      const float bias = b_proj[col];
#pragma unroll
      for (int r = 0; r < 8; ++r) {
        const int m = m0 + wm * 64 + im * 16 + r + 8 * g;
        out[(size_t)m * 1024 + col] = acc[im][in][r] + bias;
      }
    }
}

// ---------- launcher ----------
extern "C" void kernel_launch(void* const* d_in, const int* in_sizes, int n_in,
                              void* d_out, int out_size, void* d_ws, size_t ws_size,
                              hipStream_t stream) {
  const float* x      = (const float*)d_in[0];
  const float* w_attn = (const float*)d_in[1];
  const float* b_attn = (const float*)d_in[2];
  const float* w_proj = (const float*)d_in[3];
  const float* b_proj = (const float*)d_in[4];
  float* out = (float*)d_out;

  char* ws = (char*)d_ws;
  ushort_t* xb  = (ushort_t*)(ws);                        // 8 MB  [4096][1024]
  ushort_t* wab = (ushort_t*)(ws + ((size_t)8  << 20));   // 6 MB  [3072][1024] (w_attn^T)
  ushort_t* wpb = (ushort_t*)(ws + ((size_t)14 << 20));   // 2 MB  [1024][1024] (w_proj^T)
  ushort_t* qb  = (ushort_t*)(ws + ((size_t)16 << 20));   // 8 MB  [B,nh,T,hs]
  ushort_t* kb  = (ushort_t*)(ws + ((size_t)24 << 20));   // 8 MB  [B,nh,T,hs]
  ushort_t* vtb = (ushort_t*)(ws + ((size_t)32 << 20));   // 8 MB  [B,nh,hs,T]
  ushort_t* yb  = (ushort_t*)(ws + ((size_t)40 << 20));   // 8 MB  [4096][1024]

  k_cvt_bf16<<<(4096 * 1024) / 256, 256, 0, stream>>>(x, xb, 4096 * 1024);
  k_cvt_tr<<<(1024 * 3072) / 256, 256, 0, stream>>>(w_attn, wab, 1024, 3072);
  k_cvt_tr<<<(1024 * 1024) / 256, 256, 0, stream>>>(w_proj, wpb, 1024, 1024);
  k_gemm_qkv<<<dim3(16, 48), 256, 0, stream>>>(xb, wab, b_attn, qb, kb, vtb);
  k_attn<<<dim3(16, 16, 2), 256, 0, stream>>>(qb, kb, vtb, yb);
  k_gemm_proj<<<dim3(16, 16), 256, 0, stream>>>(yb, wpb, b_proj, out);
}
